// Net_1632087572625
// MI455X (gfx1250) — matrix operand order repaired and checked by reading, not compiled, not run on hardware
//
#include <hip/hip_runtime.h>
#include <hip/hip_bf16.h>
#include <math.h>

// ---------------------------------------------------------------------------
// SplineConv GNN forward for MI455X (gfx1250, wave32).
// Design: all intermediates kept < 192MB L2 (total ws ~40MB). Edge messages
// computed per-edge with a basis-combined weight (avoids the 320MB x_proj
// tensor of the reference, which would spill L2). Dense shared-B GEMMs
// (layer-2 root term, MLP layer 1) run on v_wmma_f32_16x16x32_f16 with f32
// accumulation. Aggregation via L2-resident f32 atomics. Hot edge kernels use
// b128 (float4) loads.
// ---------------------------------------------------------------------------

typedef __attribute__((ext_vector_type(16))) _Float16 v16h;
typedef __attribute__((ext_vector_type(8)))  float    v8f;

#define KDIM 5   // kernel size per dim

__device__ __forceinline__ float elu1(float v) {
    return v > 0.0f ? v : expm1f(v);
}

// degree-1 open B-spline, dim=2: 4 nonzero (slot, weight) pairs
__device__ __forceinline__ void spline4(float px, float py, int* ki, float* kw) {
    float vx = px * (float)(KDIM - 1);
    float vy = py * (float)(KDIM - 1);
    int ix = (int)floorf(vx); ix = ix < 0 ? 0 : (ix > KDIM - 2 ? KDIM - 2 : ix);
    int iy = (int)floorf(vy); iy = iy < 0 ? 0 : (iy > KDIM - 2 ? KDIM - 2 : iy);
    float fx = vx - (float)ix, fy = vy - (float)iy;
    float gx = 1.0f - fx, gy = 1.0f - fy;
    ki[0] = ix     + iy * KDIM;       kw[0] = gx * gy;   // flat idx = dim0 + dim1*K
    ki[1] = ix + 1 + iy * KDIM;       kw[1] = fx * gy;
    ki[2] = ix     + (iy + 1) * KDIM; kw[2] = gx * fy;
    ki[3] = ix + 1 + (iy + 1) * KDIM; kw[3] = fx * fy;
}

// ---------------------------------------------------------------------------
__global__ void zero_f32(float* __restrict__ p, size_t n) {
    size_t i = (size_t)blockIdx.x * blockDim.x + threadIdx.x;
    size_t stride = (size_t)gridDim.x * blockDim.x;
    for (; i < n; i += stride) p[i] = 0.0f;
}

// Layer 1 edges: Cin=1, msg = x[src] * (sum_b basis_b * W1[wi_b][:]).
// Also counts in-degree (shared with layer 2). W1 rows read as float4.
__global__ void l1_edges(const float* __restrict__ x, const int* __restrict__ ei,
                         const float* __restrict__ pseudo, const float* __restrict__ W1,
                         float* __restrict__ agg1, float* __restrict__ deg, int E) {
    int e = blockIdx.x * blockDim.x + threadIdx.x;
    if (e >= E) return;
    int s = ei[e];
    int d = ei[E + e];
    int ki[4]; float kw[4];
    spline4(pseudo[2 * e], pseudo[2 * e + 1], ki, kw);
    float xv = x[s];
    const float4* w0 = (const float4*)(W1 + (size_t)ki[0] * 32);
    const float4* w1 = (const float4*)(W1 + (size_t)ki[1] * 32);
    const float4* w2 = (const float4*)(W1 + (size_t)ki[2] * 32);
    const float4* w3 = (const float4*)(W1 + (size_t)ki[3] * 32);
    float* ap = agg1 + (size_t)d * 32;
    #pragma unroll
    for (int q = 0; q < 8; ++q) {
        float4 a0 = w0[q], a1 = w1[q], a2 = w2[q], a3 = w3[q];
        float wx = kw[0] * a0.x + kw[1] * a1.x + kw[2] * a2.x + kw[3] * a3.x;
        float wy = kw[0] * a0.y + kw[1] * a1.y + kw[2] * a2.y + kw[3] * a3.y;
        float wz = kw[0] * a0.z + kw[1] * a1.z + kw[2] * a2.z + kw[3] * a3.z;
        float ww = kw[0] * a0.w + kw[1] * a1.w + kw[2] * a2.w + kw[3] * a3.w;
        atomicAdd(&ap[q * 4 + 0], xv * wx);
        atomicAdd(&ap[q * 4 + 1], xv * wy);
        atomicAdd(&ap[q * 4 + 2], xv * wz);
        atomicAdd(&ap[q * 4 + 3], xv * ww);
    }
    atomicAdd(&deg[d], 1.0f);
}

// h1 = elu(agg1/deg + x*root1 + b1)
__global__ void l1_node(const float* __restrict__ x, const float* __restrict__ agg1,
                        const float* __restrict__ deg, const float* __restrict__ root1,
                        const float* __restrict__ b1, float* __restrict__ h1, int N) {
    int t = blockIdx.x * blockDim.x + threadIdx.x;
    if (t >= N * 32) return;
    int n = t >> 5, c = t & 31;
    float dg = deg[n]; dg = dg < 1.0f ? 1.0f : dg;
    float v = agg1[t] / dg + x[n] * root1[c] + b1[c];
    h1[t] = elu1(v);
}

// Layer 2 edges: msg = h1[src] @ Weff, Weff = sum_b basis_b * W2[wi_b].
// Thread = (edge, 16-channel group). h1 (6.4MB) and W2 (0.2MB) are L2/L1 hot.
// All weight/feature traffic uses b128 loads.
__global__ void l2_edges(const float* __restrict__ h1, const int* __restrict__ ei,
                         const float* __restrict__ pseudo, const float* __restrict__ W2,
                         float* __restrict__ agg2, int E) {
    int t = blockIdx.x * blockDim.x + threadIdx.x;
    if (t >= E * 4) return;
    int e  = t >> 2;
    int c0 = (t & 3) * 16;
    int s = ei[e];
    int d = ei[E + e];
    int ki[4]; float kw[4];
    spline4(pseudo[2 * e], pseudo[2 * e + 1], ki, kw);
    const float4* hp = (const float4*)(h1 + (size_t)s * 32);   // 8 x float4
    __builtin_prefetch(hp, 0, 1);   // global_prefetch_b8 (gfx1250)
    // W2[k][cin][cout] flat = k*2048 + cin*64 + cout; per cin: 16 floats = 4 float4
    const float4* w0 = (const float4*)(W2 + (size_t)ki[0] * 2048 + c0);
    const float4* w1 = (const float4*)(W2 + (size_t)ki[1] * 2048 + c0);
    const float4* w2 = (const float4*)(W2 + (size_t)ki[2] * 2048 + c0);
    const float4* w3 = (const float4*)(W2 + (size_t)ki[3] * 2048 + c0);
    float acc[16];
    #pragma unroll
    for (int j = 0; j < 16; ++j) acc[j] = 0.0f;
    #pragma unroll 2
    for (int q = 0; q < 8; ++q) {        // cin in chunks of 4
        float4 h4 = hp[q];
        float hv[4] = {h4.x, h4.y, h4.z, h4.w};
        #pragma unroll
        for (int u = 0; u < 4; ++u) {
            int cin = q * 4 + u;
            int base = cin * 16;         // float4 stride per cin = 64 floats = 16 float4
            #pragma unroll
            for (int jj = 0; jj < 4; ++jj) {
                float4 r0 = w0[base + jj];
                float4 r1 = w1[base + jj];
                float4 r2 = w2[base + jj];
                float4 r3 = w3[base + jj];
                float wxv = kw[0] * r0.x + kw[1] * r1.x + kw[2] * r2.x + kw[3] * r3.x;
                float wyv = kw[0] * r0.y + kw[1] * r1.y + kw[2] * r2.y + kw[3] * r3.y;
                float wzv = kw[0] * r0.z + kw[1] * r1.z + kw[2] * r2.z + kw[3] * r3.z;
                float wwv = kw[0] * r0.w + kw[1] * r1.w + kw[2] * r2.w + kw[3] * r3.w;
                acc[jj * 4 + 0] = fmaf(hv[u], wxv, acc[jj * 4 + 0]);
                acc[jj * 4 + 1] = fmaf(hv[u], wyv, acc[jj * 4 + 1]);
                acc[jj * 4 + 2] = fmaf(hv[u], wzv, acc[jj * 4 + 2]);
                acc[jj * 4 + 3] = fmaf(hv[u], wwv, acc[jj * 4 + 3]);
            }
        }
    }
    float* ap = agg2 + (size_t)d * 64 + c0;
    #pragma unroll
    for (int j = 0; j < 16; ++j) atomicAdd(&ap[j], acc[j]);
}

// Layer-2 root term: xr2[N,64] = h1[N,32] @ root2[32,64], one wave per 16x16
// D tile, v_wmma_f32_16x16x32_f16 (f16 in, f32 accumulate). Register layouts
// per ISA 7.12.2:  A: k0=(v/4)*16+half*8+(v%4)*2 ; B: kb=(v/4)*16+half*8+(v%4)*2 ;
// C/D: row = v + 8*half, col = lane%16. Full tiles (the only case when
// N%16==0) take the unguarded store path; EXEC stays all-1s around the WMMA.
__global__ __launch_bounds__(32) void xr2_wmma(const float* __restrict__ h1,
                                               const float* __restrict__ root2,
                                               float* __restrict__ xr2, int N) {
    int bid  = blockIdx.x;
    int tile = bid >> 2;       // node tile (16 rows)
    int ct   = bid & 3;        // 16-col tile of 64
    int lane = threadIdx.x;
    int r    = lane & 15;
    int half = lane >> 4;
    int col  = ct * 16 + r;
    int m    = tile * 16 + r;
    int ml   = m < N ? m : N - 1;   // clamp loads (no-op for full tiles)
    v16h a, b;
    #pragma unroll
    for (int v = 0; v < 8; ++v) {
        int ka = (v >> 2) * 16 + half * 8 + (v & 3) * 2;
        a[2 * v]     = (_Float16)h1[(size_t)ml * 32 + ka];
        a[2 * v + 1] = (_Float16)h1[(size_t)ml * 32 + ka + 1];
        int kb = (v >> 2) * 16 + half * 8 + (v & 3) * 2;
        b[2 * v]     = (_Float16)root2[kb * 64 + col];
        b[2 * v + 1] = (_Float16)root2[(kb + 1) * 64 + col];
    }
    v8f c = {};
    c = __builtin_amdgcn_wmma_f32_16x16x32_f16(false, a, false, b, (short)0, c, false, false);
    float* op = xr2 + ((size_t)tile * 16 + 8 * half) * 64 + col;
    if (tile * 16 + 15 < N) {                 // full tile: clean strided stores
        #pragma unroll
        for (int v = 0; v < 8; ++v) op[(size_t)v * 64] = c[v];
    } else {                                  // tail tile (not hit when N%16==0)
        #pragma unroll
        for (int v = 0; v < 8; ++v) {
            int row = tile * 16 + v + 8 * half;
            if (row < N) xr2[(size_t)row * 64 + col] = c[v];
        }
    }
}

// h2 = elu(agg2/deg + xr2 + b2), fused with global_mean_pool accumulation.
__global__ void l2_node_pool(const float* __restrict__ agg2, const float* __restrict__ deg,
                             const float* __restrict__ xr2, const float* __restrict__ b2,
                             const int* __restrict__ batch, float* __restrict__ pooled,
                             float* __restrict__ cnt, int N) {
    int t = blockIdx.x * blockDim.x + threadIdx.x;
    if (t >= N * 64) return;
    int n = t >> 6, c = t & 63;
    float dg = deg[n]; dg = dg < 1.0f ? 1.0f : dg;
    float v = agg2[t] / dg + xr2[t] + b2[c];
    float h = elu1(v);
    int g = batch[n];
    atomicAdd(&pooled[(size_t)g * 64 + c], h);
    if (c == 0) atomicAdd(&cnt[g], 1.0f);
}

__global__ void pool_fin(const float* __restrict__ pooled, const float* __restrict__ cnt,
                         float* __restrict__ pm, int G) {
    int t = blockIdx.x * blockDim.x + threadIdx.x;
    if (t >= G * 64) return;
    int g = t >> 6;
    float cc = cnt[g]; cc = cc < 1.0f ? 1.0f : cc;
    pm[t] = pooled[t] / cc;
}

// MLP layer 1: g1[G,128] = elu(pm[G,64] @ Wl1[64,128] + bl1). K=64 -> two
// 16x16x32 WMMA steps accumulating into one f32 tile. G is a multiple of 16.
__global__ __launch_bounds__(32) void mlp1_wmma(const float* __restrict__ pm,
                                                const float* __restrict__ Wl1,
                                                const float* __restrict__ bl1,
                                                float* __restrict__ g1) {
    int bid  = blockIdx.x;
    int rt   = bid >> 3;       // row tile of G
    int ct   = bid & 7;        // col tile of 128
    int lane = threadIdx.x;
    int r    = lane & 15;
    int half = lane >> 4;
    int col  = ct * 16 + r;
    int m    = rt * 16 + r;
    v8f c = {};
    #pragma unroll
    for (int kk = 0; kk < 64; kk += 32) {
        v16h a, b;
        #pragma unroll
        for (int v = 0; v < 8; ++v) {
            int ka = kk + (v >> 2) * 16 + half * 8 + (v & 3) * 2;
            a[2 * v]     = (_Float16)pm[(size_t)m * 64 + ka];
            a[2 * v + 1] = (_Float16)pm[(size_t)m * 64 + ka + 1];
            int kb = kk + (v >> 2) * 16 + half * 8 + (v & 3) * 2;
            b[2 * v]     = (_Float16)Wl1[kb * 128 + col];
            b[2 * v + 1] = (_Float16)Wl1[(kb + 1) * 128 + col];
        }
        c = __builtin_amdgcn_wmma_f32_16x16x32_f16(false, a, false, b, (short)0, c, false, false);
    }
    float* op = g1 + ((size_t)rt * 16 + 8 * half) * 128 + col;
    float bv = bl1[col];
    #pragma unroll
    for (int v = 0; v < 8; ++v) op[(size_t)v * 128] = elu1(c[v] + bv);
}

// logits = g1 @ Wl2 + bl2, then log_softmax over 10 classes. One block/graph.
__global__ __launch_bounds__(32) void mlp2_logsoftmax(const float* __restrict__ g1,
                                                      const float* __restrict__ Wl2,
                                                      const float* __restrict__ bl2,
                                                      float* __restrict__ out) {
    __shared__ float l[10];
    int g = blockIdx.x, t = threadIdx.x;
    if (t < 10) {
        float acc = bl2[t];
        const float* gp = g1 + (size_t)g * 128;
        for (int k = 0; k < 128; ++k) acc = fmaf(gp[k], Wl2[k * 10 + t], acc);
        l[t] = acc;
    }
    __syncthreads();
    if (t < 10) {
        float m = l[0];
        #pragma unroll
        for (int i = 1; i < 10; ++i) m = fmaxf(m, l[i]);
        float s = 0.0f;
        #pragma unroll
        for (int i = 0; i < 10; ++i) s += expf(l[i] - m);
        out[g * 10 + t] = l[t] - m - logf(s);
    }
}

// ---------------------------------------------------------------------------
extern "C" void kernel_launch(void* const* d_in, const int* in_sizes, int n_in,
                              void* d_out, int out_size, void* d_ws, size_t ws_size,
                              hipStream_t stream) {
    const float* x      = (const float*)d_in[0];   // [N,1]
    const int*   ei     = (const int*)d_in[1];     // [2,E]
    const float* pseudo = (const float*)d_in[2];   // [E,2]
    const int*   batch  = (const int*)d_in[3];     // [N]
    const float* W1     = (const float*)d_in[4];   // [25,1,32]
    const float* root1  = (const float*)d_in[5];   // [1,32]
    const float* b1     = (const float*)d_in[6];   // [32]
    const float* W2     = (const float*)d_in[7];   // [25,32,64]
    const float* root2  = (const float*)d_in[8];   // [32,64]
    const float* b2     = (const float*)d_in[9];   // [64]
    const float* Wl1    = (const float*)d_in[10];  // [64,128]
    const float* bl1    = (const float*)d_in[11];  // [128]
    const float* Wl2    = (const float*)d_in[12];  // [128,10]
    const float* bl2    = (const float*)d_in[13];  // [10]
    float* out = (float*)d_out;

    const int N = in_sizes[0];
    const int E = in_sizes[1] / 2;
    const int G = out_size / 10;

    // Workspace layout (floats). Accumulators (zeroed each call) first.
    float* ws     = (float*)d_ws;
    float* agg1   = ws;                              // N*32
    float* agg2   = agg1 + (size_t)N * 32;           // N*64
    float* deg    = agg2 + (size_t)N * 64;           // N
    float* pooled = deg + N;                         // G*64
    float* cnt    = pooled + (size_t)G * 64;         // G
    size_t zeroCount = (size_t)N * 97 + (size_t)G * 65;
    float* h1     = cnt + G;                         // N*32
    float* xr2    = h1 + (size_t)N * 32;             // N*64
    float* pm     = xr2 + (size_t)N * 64;            // G*64
    float* g1     = pm + (size_t)G * 64;             // G*128
    (void)ws_size; (void)n_in;

    const int B = 256;
    zero_f32<<<4096, B, 0, stream>>>(ws, zeroCount);

    l1_edges<<<(E + B - 1) / B, B, 0, stream>>>(x, ei, pseudo, W1, agg1, deg, E);
    l1_node<<<((size_t)N * 32 + B - 1) / B, B, 0, stream>>>(x, agg1, deg, root1, b1, h1, N);

    l2_edges<<<((size_t)E * 4 + B - 1) / B, B, 0, stream>>>(h1, ei, pseudo, W2, agg2, E);
    xr2_wmma<<<((N + 15) / 16) * 4, 32, 0, stream>>>(h1, root2, xr2, N);
    l2_node_pool<<<((size_t)N * 64 + B - 1) / B, B, 0, stream>>>(agg2, deg, xr2, b2, batch,
                                                                 pooled, cnt, N);

    pool_fin<<<((size_t)G * 64 + B - 1) / B, B, 0, stream>>>(pooled, cnt, pm, G);
    mlp1_wmma<<<((G + 15) / 16) * 8, 32, 0, stream>>>(pm, Wl1, bl1, g1);
    mlp2_logsoftmax<<<G, 32, 0, stream>>>(g1, Wl2, bl2, out);
}
